// ParallelLlamaRingAttention_17291538333985
// MI455X (gfx1250) — matrix operand order repaired and checked by reading, not compiled, not run on hardware
//
#include <hip/hip_runtime.h>
#include <hip/hip_bf16.h>

// ---------------- CDNA5 (gfx1250) types ----------------
typedef __bf16 bf16_t;
typedef __attribute__((ext_vector_type(16))) __bf16 v16bf;
typedef __attribute__((ext_vector_type(8)))  __bf16 v8bf;
typedef __attribute__((ext_vector_type(8)))  float  v8f;

// ---------------- Problem dims ----------------
#define S_LEN   2048
#define BATCH   2
#define DMODEL  4096
#define NHEADS  32
#define NKVH    8
#define HEADD   128
#define QKVC    6144                 // (NH + 2*NKV) * HD
#define TOKENS  (S_LEN * BATCH)      // 4096
#define KOFF    (NHEADS * HEADD)     // 4096 : K column offset in QKV
#define VOFF    ((NHEADS + NKVH) * HEADD) // 5120 : V column offset

__device__ __forceinline__ v16bf cat8(v8bf lo, v8bf hi) {
    return __builtin_shufflevector(lo, hi, 0,1,2,3,4,5,6,7,8,9,10,11,12,13,14,15);
}

__device__ __forceinline__ v8f wmma_bf16(v16bf a, v16bf b, v8f c) {
    return __builtin_amdgcn_wmma_f32_16x16x32_bf16(
        /*neg_a=*/false, a, /*neg_b=*/false, b,
        /*c_mod=*/(short)0, c, /*reuse_a=*/false, /*reuse_b=*/false);
}

// A fragment (16x32 bf16): lane<16 -> row M=lane, K 2*klo..2*klo+15 contiguous
// klo = (lane>>4)*8. base points at A[row] + k0.
__device__ __forceinline__ v16bf load_a_frag(const bf16_t* base, int klo) {
    v8bf lo = *reinterpret_cast<const v8bf*>(base + 2 * klo);
    v8bf hi = *reinterpret_cast<const v8bf*>(base + 2 * klo + 8);
    return cat8(lo, hi);
}

// B fragment (32x16 bf16): lane holds column n=lane&15, K kb..kb+15 contiguous
// (from a [N,K]-transposed operand). base points at Bt[n] + k0 + kb.
__device__ __forceinline__ v16bf load_b_frag(const bf16_t* base) {
    v8bf lo = *reinterpret_cast<const v8bf*>(base);
    v8bf hi = *reinterpret_cast<const v8bf*>(base + 8);
    return cat8(lo, hi);
}

// ---------------- prep kernels ----------------
__global__ void cvt_f32_to_bf16(const float* __restrict__ in,
                                bf16_t* __restrict__ out, size_t n) {
    size_t i = (size_t)blockIdx.x * blockDim.x + threadIdx.x;
    size_t stride = (size_t)gridDim.x * blockDim.x;
    for (; i < n; i += stride) out[i] = (bf16_t)in[i];
}

// in: [K,N] f32  ->  out: [N,K] bf16 (output-index coalesced)
__global__ void transpose_f32_to_bf16(const float* __restrict__ in,
                                      bf16_t* __restrict__ out, int K, int N) {
    size_t total = (size_t)K * N;
    size_t i = (size_t)blockIdx.x * blockDim.x + threadIdx.x;
    size_t stride = (size_t)gridDim.x * blockDim.x;
    for (; i < total; i += stride) {
        int nn = (int)(i / K);
        int kk = (int)(i % K);
        out[i] = (bf16_t)in[(size_t)kk * N + nn];
    }
}

// Vt[b][kvh][d][s] = QKV[(s*B+b)][VOFF + kvh*HD + d]
__global__ void extract_vt(const bf16_t* __restrict__ qkv,
                           bf16_t* __restrict__ vt) {
    size_t total = (size_t)BATCH * NKVH * HEADD * S_LEN;
    size_t i = (size_t)blockIdx.x * blockDim.x + threadIdx.x;
    size_t stride = (size_t)gridDim.x * blockDim.x;
    for (; i < total; i += stride) {
        int s   = (int)(i % S_LEN);
        size_t t = i / S_LEN;
        int d   = (int)(t % HEADD); t /= HEADD;
        int kvh = (int)(t % NKVH);
        int b   = (int)(t / NKVH);
        vt[i] = qkv[(size_t)(s * BATCH + b) * QKVC + VOFF + kvh * HEADD + d];
    }
}

// ---------------- bf16 WMMA GEMM: C[M,N] = A[M,K] * Bt[N,K]^T ----------------
// block = 128 threads = 4 waves (2x2), each wave owns a 64x64 output tile.
template <typename OutT>
__global__ __launch_bounds__(128) void gemm_bf16_wmma(
    const bf16_t* __restrict__ A,   // [M,K] row-major
    const bf16_t* __restrict__ Bt,  // [N,K] row-major (i.e. B transposed)
    OutT* __restrict__ C,           // [M,N] row-major
    int M, int N, int K) {
    const int lane  = threadIdx.x & 31;
    const int wave  = threadIdx.x >> 5;
    const int m_base = blockIdx.y * 128 + (wave >> 1) * 64;
    const int n_base = blockIdx.x * 128 + (wave & 1) * 64;

    const int l16 = lane & 15;
    const int klo = (lane >> 4) * 8;   // A fragment K sub-offset
    const int kb  = (lane >> 4) * 16;  // B fragment K sub-offset

    v8f acc[4][4];
#pragma unroll
    for (int i = 0; i < 4; i++)
#pragma unroll
        for (int j = 0; j < 4; j++)
#pragma unroll
            for (int e = 0; e < 8; e++) acc[i][j][e] = 0.f;

    const bf16_t* arow[4];
    const bf16_t* brow[4];
#pragma unroll
    for (int mi = 0; mi < 4; mi++)
        arow[mi] = A + (size_t)(m_base + mi * 16 + l16) * K;
#pragma unroll
    for (int ni = 0; ni < 4; ni++)
        brow[ni] = Bt + (size_t)(n_base + ni * 16 + l16) * K;

    for (int k0 = 0; k0 < K; k0 += 32) {
        if (k0 + 64 < K) {  // prefetch next-next K tile -> global_prefetch_b8
            __builtin_prefetch((const void*)(arow[0] + k0 + 64), 0, 1);
            __builtin_prefetch((const void*)(brow[0] + k0 + 64), 0, 1);
        }
        v16bf a[4], b[4];
#pragma unroll
        for (int mi = 0; mi < 4; mi++) a[mi] = load_a_frag(arow[mi] + k0, klo);
#pragma unroll
        for (int ni = 0; ni < 4; ni++) b[ni] = load_b_frag(brow[ni] + k0 + kb);
#pragma unroll
        for (int mi = 0; mi < 4; mi++)
#pragma unroll
            for (int ni = 0; ni < 4; ni++)
                acc[mi][ni] = wmma_bf16(a[mi], b[ni], acc[mi][ni]);
    }

    // C layout: value (r, lane) -> row = r + 8*(lane>>4), col = lane&15
    const int rhalf = (lane >> 4) * 8;
#pragma unroll
    for (int mi = 0; mi < 4; mi++)
#pragma unroll
        for (int ni = 0; ni < 4; ni++) {
            int col = n_base + ni * 16 + l16;
#pragma unroll
            for (int r = 0; r < 8; r++) {
                int row = m_base + mi * 16 + r + rhalf;
                C[(size_t)row * N + col] = (OutT)acc[mi][ni][r];
            }
        }
}

// ---------------- flash attention (GQA, non-causal) ----------------
// grid = (S/64, NH, B), block = 128 (4 waves); wave w -> query tile of 16 rows.
__global__ __launch_bounds__(128) void flash_attn_wmma(
    const bf16_t* __restrict__ QKV,  // [TOKENS, QKVC] bf16
    const bf16_t* __restrict__ Vt,   // [B][NKV][HD][S] bf16
    bf16_t* __restrict__ attn) {     // [TOKENS, NH*HD] bf16
    __shared__ __align__(16) bf16_t plds[4][16][32];

    const int lane = threadIdx.x & 31;
    const int wave = threadIdx.x >> 5;
    const int h   = blockIdx.y;
    const int b   = blockIdx.z;
    const int kvh = h >> 2;               // NH/NKV = 4 groups
    const int q0  = (blockIdx.x * 4 + wave) * 16;

    const int l16  = lane & 15;
    const int half = lane >> 4;           // 0: rows 0-7 / 1: rows 8-15
    const int klo  = half * 8;
    const int kb   = half * 16;
    const float scale = 0.088388347648318447f;  // 1/sqrt(128)

    // Q fragments over HD=128 (4 chunks of K=32)
    v16bf qf[4];
    {
        const bf16_t* qrow =
            QKV + (size_t)((q0 + l16) * BATCH + b) * QKVC + h * HEADD;
#pragma unroll
        for (int kc = 0; kc < 4; kc++) qf[kc] = load_a_frag(qrow + kc * 32, klo);
    }

    float m[8], l[8];
    v8f acc[8];
#pragma unroll
    for (int r = 0; r < 8; r++) { m[r] = -1e30f; l[r] = 0.f; }
#pragma unroll
    for (int di = 0; di < 8; di++)
#pragma unroll
        for (int e = 0; e < 8; e++) acc[di][e] = 0.f;

    for (int t0 = 0; t0 < S_LEN; t0 += 32) {
        // ---- S = Q * K^T for 32 keys (two 16x16 tiles) ----
        v8f s0, s1;
#pragma unroll
        for (int e = 0; e < 8; e++) { s0[e] = 0.f; s1[e] = 0.f; }
#pragma unroll
        for (int t = 0; t < 2; t++) {
            const bf16_t* krow =
                QKV + (size_t)((t0 + t * 16 + l16) * BATCH + b) * QKVC +
                KOFF + kvh * HEADD;
#pragma unroll
            for (int kc = 0; kc < 4; kc++) {
                v16bf kf = load_b_frag(krow + kc * 32 + kb);
                if (t == 0) s0 = wmma_bf16(qf[kc], kf, s0);
                else        s1 = wmma_bf16(qf[kc], kf, s1);
            }
        }
        // ---- online softmax over the 32 columns ----
        float alpha[8];
#pragma unroll
        for (int r = 0; r < 8; r++) {
            float v0 = s0[r] * scale, v1 = s1[r] * scale;
            float mx = fmaxf(v0, v1);
#pragma unroll
            for (int off = 1; off < 16; off <<= 1)
                mx = fmaxf(mx, __shfl_xor(mx, off, 32));
            float mnew = fmaxf(m[r], mx);
            float p0 = __expf(v0 - mnew);
            float p1 = __expf(v1 - mnew);
            float rs = p0 + p1;
#pragma unroll
            for (int off = 1; off < 16; off <<= 1)
                rs += __shfl_xor(rs, off, 32);
            alpha[r] = __expf(m[r] - mnew);
            l[r] = l[r] * alpha[r] + rs;
            m[r] = mnew;
            s0[r] = p0; s1[r] = p1;
        }
        // ---- re-shape P (C layout) -> A layout via LDS ----
#pragma unroll
        for (int r = 0; r < 8; r++) {
            int row = r + 8 * half;
            plds[wave][row][l16]      = (bf16_t)s0[r];
            plds[wave][row][16 + l16] = (bf16_t)s1[r];
        }
        const bf16_t* prow = &plds[wave][l16][0];
        v16bf pf = cat8(*reinterpret_cast<const v8bf*>(prow + 2 * klo),
                        *reinterpret_cast<const v8bf*>(prow + 2 * klo + 8));
        // ---- acc = acc*alpha + P @ V ----
        const bf16_t* vbase =
            Vt + ((size_t)(b * NKVH + kvh) * HEADD) * S_LEN + t0 + kb;
#pragma unroll
        for (int di = 0; di < 8; di++) {
#pragma unroll
            for (int e = 0; e < 8; e++) acc[di][e] = acc[di][e] * alpha[e];
            const bf16_t* vrow = vbase + (size_t)(di * 16 + l16) * S_LEN;
            v16bf vf = load_b_frag(vrow);
            acc[di] = wmma_bf16(pf, vf, acc[di]);
        }
    }

    // ---- normalize + store: attn[(s*B+b)][h*HD + d] ----
#pragma unroll
    for (int r = 0; r < 8; r++) {
        float rl = 1.f / l[r];
        int srow = q0 + r + 8 * half;
#pragma unroll
        for (int di = 0; di < 8; di++) {
            int col = h * HEADD + di * 16 + l16;
            attn[(size_t)(srow * BATCH + b) * (NHEADS * HEADD) + col] =
                (bf16_t)(acc[di][r] * rl);
        }
    }
}

// ---------------- host side ----------------
extern "C" void kernel_launch(void* const* d_in, const int* in_sizes, int n_in,
                              void* d_out, int out_size, void* d_ws,
                              size_t ws_size, hipStream_t stream) {
    const float* hidden = (const float*)d_in[0];  // [S,B,D] f32
    const float* w_qkv  = (const float*)d_in[1];  // [D, QKVC] f32
    const float* w_o    = (const float*)d_in[2];  // [D, D] f32
    float* out = (float*)d_out;                   // [S,B,D] f32

    char* ws = (char*)d_ws;
    size_t off = 0;
    bf16_t* Xb     = (bf16_t*)(ws + off); off += (size_t)TOKENS * DMODEL * 2;
    bf16_t* WqkvT  = (bf16_t*)(ws + off); off += (size_t)QKVC * DMODEL * 2;
    bf16_t* WoT    = (bf16_t*)(ws + off); off += (size_t)DMODEL * DMODEL * 2;
    bf16_t* QKVb   = (bf16_t*)(ws + off); off += (size_t)TOKENS * QKVC * 2;
    bf16_t* Vt     = (bf16_t*)(ws + off); off += (size_t)BATCH * NKVH * HEADD * S_LEN * 2;
    bf16_t* attn   = (bf16_t*)(ws + off);

    // 1) convert activations / transpose+convert weights to bf16
    {
        size_t n = (size_t)TOKENS * DMODEL;
        cvt_f32_to_bf16<<<(int)((n + 255) / 256), 256, 0, stream>>>(hidden, Xb, n);
    }
    {
        size_t n = (size_t)DMODEL * QKVC;
        transpose_f32_to_bf16<<<(int)((n + 255) / 256), 256, 0, stream>>>(
            w_qkv, WqkvT, DMODEL, QKVC);
    }
    {
        size_t n = (size_t)DMODEL * DMODEL;
        transpose_f32_to_bf16<<<(int)((n + 255) / 256), 256, 0, stream>>>(
            w_o, WoT, DMODEL, DMODEL);
    }
    // 2) fused QKV projection (bf16 WMMA), bf16 output
    gemm_bf16_wmma<bf16_t><<<dim3(QKVC / 128, TOKENS / 128), 128, 0, stream>>>(
        Xb, WqkvT, QKVb, TOKENS, QKVC, DMODEL);
    // 3) V transpose for contiguous P@V B-fragments
    {
        size_t n = (size_t)BATCH * NKVH * HEADD * S_LEN;
        extract_vt<<<(int)((n + 255) / 256), 256, 0, stream>>>(QKVb, Vt);
    }
    // 4) flash attention (online softmax over 4 ring blocks == full seq scan)
    flash_attn_wmma<<<dim3(S_LEN / 64, NHEADS, BATCH), 128, 0, stream>>>(
        QKVb, Vt, attn);
    // 5) output projection, f32 result straight into d_out
    gemm_bf16_wmma<float><<<dim3(DMODEL / 128, TOKENS / 128), 128, 0, stream>>>(
        attn, WoT, out, TOKENS, DMODEL, DMODEL);
}
